// WorkingMemory_78176994722415
// MI455X (gfx1250) — matrix operand order repaired and checked by reading, not compiled, not run on hardware
//
#include <hip/hip_runtime.h>
#include <hip/hip_bf16.h>

#define B_ 32
#define S_ 4096
#define H_ 256
#define MEM_ 256
#define HEADS_ 4
#define HD_ 64
#define LDP 264        // padded LDS row stride (elements); 264*2B = 33*16B -> 16B aligned rows
#define TILE_E 512     // packed B-tile: 32 lanes * 16 bf16

typedef __attribute__((ext_vector_type(16))) __bf16 v16bf;
typedef __attribute__((ext_vector_type(8)))  __bf16 v8bf;
typedef __attribute__((ext_vector_type(8)))  float  v8f;

// ---- WMMA fragment helpers (CDNA5 16x16x32 bf16 layouts, ISA 7.12.2) ----

// A 16x32: lane&15 = row M, lane>>4 = K-half. Per lane the 16 values are one
// contiguous 16-elem run: [kbase+half*16 .. +16), as two 8-elem chunks.
// Both 16B-aligned (ld=264, kbase%32==0) -> two ds_load_b128.
__device__ inline v16bf lds_load_a(const __bf16* base, int ld, int kbase) {
  int lane = threadIdx.x & 31;
  int row = lane & 15, half = lane >> 4;
  const __bf16* p = base + row * ld + kbase + half * 16;
  v8bf lo = *(const v8bf*)p;
  v8bf hi = *(const v8bf*)(p + 8);
  return __builtin_shufflevector(lo, hi, 0, 1, 2, 3, 4, 5, 6, 7, 8, 9, 10, 11, 12,
                                 13, 14, 15);
}

// B 32x16 from fragment-packed global tile: lane's 16 bf16 contiguous (32B)
// -> two global_load_b128.
__device__ inline v16bf load_b_packed(const __bf16* tilebase) {
  int lane = threadIdx.x & 31;
  const v8bf* p = (const v8bf*)(tilebase + (size_t)lane * 16);
  v8bf lo = p[0];
  v8bf hi = p[1];
  return __builtin_shufflevector(lo, hi, 0, 1, 2, 3, 4, 5, 6, 7, 8, 9, 10, 11, 12,
                                 13, 14, 15);
}

__device__ inline v8f wmma_bf16(v16bf a, v16bf b, v8f c) {
  // (neg_a, A, neg_b, B, c_mod, C, reuse_a, reuse_b)
  return __builtin_amdgcn_wmma_f32_16x16x32_bf16(false, a, false, b, (short)0, c,
                                                 false, false);
}

// ---------------------------------------------------------------- kernels

__global__ void zero_kernel(float* p, int n) {
  int i = blockIdx.x * blockDim.x + threadIdx.x;
  if (i < n) p[i] = 0.f;
}

// Re-pack f32 weights into fragment-packed bf16:
// packed[((ktile*16+ntile)*32 + lane)*16 + i] = W[(ktile*32+(lane>>4)*16+i)*256 + ntile*16+(lane&15)]
__global__ __launch_bounds__(256) void pack_weights_kernel(
    const float* __restrict__ Wq, const float* __restrict__ Wk,
    const float* __restrict__ Wv, const float* __restrict__ Wo,
    __bf16* __restrict__ Wq_p, __bf16* __restrict__ Wk_p,
    __bf16* __restrict__ Wv_p, __bf16* __restrict__ Wo_p) {
  int run = blockIdx.x * blockDim.x + threadIdx.x;  // one 16-element lane-run
  const float* src;
  __bf16* dst;
  int r;
  if (run < 4096) {           // Wq: 256x256 -> 4096 runs
    src = Wq; dst = Wq_p; r = run;
  } else if (run < 8192) {    // Wk
    src = Wk; dst = Wk_p; r = run - 4096;
  } else if (run < 12288) {   // Wv
    src = Wv; dst = Wv_p; r = run - 8192;
  } else if (run < 20480) {   // Wo: 512x256 -> 8192 runs
    src = Wo; dst = Wo_p; r = run - 12288;
  } else {
    return;
  }
  int lane = r & 31;
  int tile = r >> 5;  // ktile*16 + ntile
  int ntile = tile & 15, ktile = tile >> 4;
  int k0 = ktile * 32 + (lane >> 4) * 16;
  int n = ntile * 16 + (lane & 15);
#pragma unroll
  for (int i = 0; i < 16; ++i)
    dst[(size_t)r * 16 + i] = (__bf16)src[(size_t)(k0 + i) * H_ + n];
}

// K = memory@Wk + bk -> packed Ktp[b][h][kk(2)][mt(16)] tiles of B[d][m]
// V = memory@Wv + bv -> packed Vbp[b][h][kk(8)][dt(4)] tiles of B[m][d]
__global__ __launch_bounds__(256) void kv_kernel(
    const float* __restrict__ memory, const float* __restrict__ bk,
    const float* __restrict__ bv, const __bf16* __restrict__ Wk_p,
    const __bf16* __restrict__ Wv_p, __bf16* __restrict__ Ktp,
    __bf16* __restrict__ Vbp) {
  __shared__ __bf16 sm[16][LDP];
  const int b = blockIdx.x >> 4;
  const int mt = blockIdx.x & 15;
  const int tid = threadIdx.x;
  const int wave = tid >> 5;
  const int row16 = tid >> 4, ln16 = tid & 15;
  {
    const float* mp =
        memory + ((size_t)b * MEM_ + mt * 16 + row16) * H_ + ln16 * 16;
#pragma unroll
    for (int jj = 0; jj < 4; ++jj) {
      float4 f = ((const float4*)mp)[jj];
      int c = ln16 * 16 + jj * 4;
      sm[row16][c + 0] = (__bf16)f.x;
      sm[row16][c + 1] = (__bf16)f.y;
      sm[row16][c + 2] = (__bf16)f.z;
      sm[row16][c + 3] = (__bf16)f.w;
    }
  }
  __syncthreads();
  const int lane = tid & 31, half = lane >> 4, cn = lane & 15;
  for (int ct = wave; ct < 16; ct += 8) {
    int colg = ct * 16 + cn;
    int hh = colg >> 6, d = colg & 63;
    {  // K (software-pipelined B fragments)
      float bias = bk[colg];
      v8f acc;
#pragma unroll
      for (int r = 0; r < 8; ++r) acc[r] = bias;
      v16bf bm = load_b_packed(Wk_p + (size_t)ct * TILE_E);
#pragma unroll
      for (int kk = 0; kk < 8; ++kk) {
        v16bf bnext = bm;
        if (kk < 7)
          bnext = load_b_packed(Wk_p + (size_t)((kk + 1) * 16 + ct) * TILE_E);
        v16bf a = lds_load_a(&sm[0][0], LDP, kk * 32);
        acc = wmma_bf16(a, bm, acc);
        bm = bnext;
      }
      int kkp = d >> 5, ip = d & 15, lhalf = (d >> 4) & 1;
#pragma unroll
      for (int r = 0; r < 8; ++r) {
        int m = mt * 16 + r + 8 * half;
        int lane_p = lhalf * 16 + (m & 15);
        Ktp[((((size_t)(b * HEADS_ + hh) * 2 + kkp) * 16 + (m >> 4)) * 32 + lane_p) *
                16 +
            ip] = (__bf16)acc[r];
      }
    }
    {  // V (software-pipelined B fragments)
      float bias = bv[colg];
      v8f acc;
#pragma unroll
      for (int r = 0; r < 8; ++r) acc[r] = bias;
      v16bf bm = load_b_packed(Wv_p + (size_t)ct * TILE_E);
#pragma unroll
      for (int kk = 0; kk < 8; ++kk) {
        v16bf bnext = bm;
        if (kk < 7)
          bnext = load_b_packed(Wv_p + (size_t)((kk + 1) * 16 + ct) * TILE_E);
        v16bf a = lds_load_a(&sm[0][0], LDP, kk * 32);
        acc = wmma_bf16(a, bm, acc);
        bm = bnext;
      }
      int dtv = d >> 4;
#pragma unroll
      for (int r = 0; r < 8; ++r) {
        int m = mt * 16 + r + 8 * half;
        int lane_v = ((m >> 4) & 1) * 16 + (d & 15);
        Vbp[((((size_t)(b * HEADS_ + hh) * 8 + (m >> 5)) * 4 + dtv) * 32 + lane_v) *
                16 +
            (m & 15)] = (__bf16)acc[r];
      }
    }
  }
}

__global__ __launch_bounds__(256) void xmean_kernel(const float* __restrict__ x,
                                                    float* __restrict__ avg) {
  int b = blockIdx.x, h = threadIdx.x;
  float s = 0.f;
  const float* xp = x + (size_t)b * S_ * H_ + h;
  for (int i = 0; i < S_; ++i) s += xp[(size_t)i * H_];
  avg[b * H_ + h] = s * (1.f / (float)S_);
}

// Fused: Q-proj -> per-head scores/softmax/ctx -> Wo -> LayerNorm -> gate -> gsum atomics
__global__ __launch_bounds__(256) void attn_kernel(
    const float* __restrict__ x, const float* __restrict__ bq,
    const float* __restrict__ bo, const float* __restrict__ Wg,
    const float* __restrict__ bg, const float* __restrict__ ln_g,
    const float* __restrict__ ln_b, const __bf16* __restrict__ Wq_p,
    const __bf16* __restrict__ Wo_p, const __bf16* __restrict__ Ktp,
    const __bf16* __restrict__ Vbp, float* __restrict__ out_ro,
    float* __restrict__ out_ma, float* __restrict__ gsum) {
  __shared__ __bf16 sx[16][LDP];
  __shared__ __bf16 sq[16][LDP];
  __shared__ __bf16 sattn[16][LDP];
  __shared__ __bf16 sctx[16][LDP];
  __shared__ float sf[16][LDP];
  __shared__ float srow[16];

  const int b = blockIdx.y;
  const int s0 = blockIdx.x * 16;
  const int tid = threadIdx.x;
  const int wave = tid >> 5;
  const int lane = tid & 31, half = lane >> 4, cn = lane & 15;
  const int row16 = tid >> 4, ln16 = tid & 15;

  // stage 0: x tile -> LDS bf16 (vectorized)
  {
    const float* xp = x + ((size_t)b * S_ + s0 + row16) * H_ + ln16 * 16;
#pragma unroll
    for (int jj = 0; jj < 4; ++jj) {
      float4 f = ((const float4*)xp)[jj];
      int c = ln16 * 16 + jj * 4;
      sx[row16][c + 0] = (__bf16)f.x;
      sx[row16][c + 1] = (__bf16)f.y;
      sx[row16][c + 2] = (__bf16)f.z;
      sx[row16][c + 3] = (__bf16)f.w;
    }
  }
  __syncthreads();

  // stage 1: Q = x@Wq + bq   (B fragments software-pipelined)
  for (int ct = wave; ct < 16; ct += 8) {
    int colg = ct * 16 + cn;
    float bias = bq[colg];
    v8f acc;
#pragma unroll
    for (int r = 0; r < 8; ++r) acc[r] = bias;
    v16bf bm = load_b_packed(Wq_p + (size_t)ct * TILE_E);
#pragma unroll
    for (int kk = 0; kk < 8; ++kk) {
      v16bf bnext = bm;
      if (kk < 7)
        bnext = load_b_packed(Wq_p + (size_t)((kk + 1) * 16 + ct) * TILE_E);
      v16bf a = lds_load_a(&sx[0][0], LDP, kk * 32);
      acc = wmma_bf16(a, bm, acc);
      bm = bnext;
    }
#pragma unroll
    for (int r = 0; r < 8; ++r) sq[r + 8 * half][colg] = (__bf16)acc[r];
  }
  __syncthreads();

  float macc[16];
#pragma unroll
  for (int j = 0; j < 16; ++j) macc[j] = 0.f;

  // stage 2: heads
  for (int h = 0; h < HEADS_; ++h) {
    const __bf16* Kh = Ktp + (size_t)(b * HEADS_ + h) * 2 * 16 * TILE_E;
    const __bf16* Vh = Vbp + (size_t)(b * HEADS_ + h) * 8 * 4 * TILE_E;
    // scores[s,m] = sum_d q[s,h*64+d] * K[m,h*64+d]
    for (int mt = wave; mt < 16; mt += 8) {
      v8f acc;
#pragma unroll
      for (int r = 0; r < 8; ++r) acc[r] = 0.f;
      v16bf b0 = load_b_packed(Kh + (size_t)mt * TILE_E);
      v16bf b1 = load_b_packed(Kh + (size_t)(16 + mt) * TILE_E);
      v16bf a0 = lds_load_a(&sq[0][0], LDP, h * HD_);
      v16bf a1 = lds_load_a(&sq[0][0], LDP, h * HD_ + 32);
      acc = wmma_bf16(a0, b0, acc);
      acc = wmma_bf16(a1, b1, acc);
#pragma unroll
      for (int r = 0; r < 8; ++r) sf[r + 8 * half][mt * 16 + cn] = acc[r];
    }
    __syncthreads();

    // row softmax (16 lanes per row, wave32 shuffle reductions)
    {
      float vloc[16];
      float mx = -3.4e38f;
#pragma unroll
      for (int j = 0; j < 16; ++j) {
        float v = sf[row16][ln16 + 16 * j] * 0.125f;  // 1/sqrt(64)
        vloc[j] = v;
        mx = fmaxf(mx, v);
      }
      for (int m = 1; m < 16; m <<= 1) mx = fmaxf(mx, __shfl_xor(mx, m, 32));
      float sm = 0.f;
#pragma unroll
      for (int j = 0; j < 16; ++j) {
        vloc[j] = __expf(vloc[j] - mx);
        sm += vloc[j];
      }
      for (int m = 1; m < 16; m <<= 1) sm += __shfl_xor(sm, m, 32);
      float inv = 1.f / sm;
#pragma unroll
      for (int j = 0; j < 16; ++j) {
        float a = vloc[j] * inv;
        sattn[row16][ln16 + 16 * j] = (__bf16)a;
        macc[j] += a * 0.25f;  // mean over 4 heads
      }
    }
    __syncthreads();

    // ctx[s,d] = sum_m attn[s,m] * V[m,h*64+d]  (4 tiles, waves 0..3)
    if (wave < 4) {
      int dt = wave;
      v8f acc;
#pragma unroll
      for (int r = 0; r < 8; ++r) acc[r] = 0.f;
      v16bf bm = load_b_packed(Vh + (size_t)dt * TILE_E);
#pragma unroll
      for (int kk = 0; kk < 8; ++kk) {
        v16bf bnext = bm;
        if (kk < 7)
          bnext = load_b_packed(Vh + (size_t)((kk + 1) * 4 + dt) * TILE_E);
        v16bf a = lds_load_a(&sattn[0][0], LDP, kk * 32);
        acc = wmma_bf16(a, bm, acc);
        bm = bnext;
      }
#pragma unroll
      for (int r = 0; r < 8; ++r)
        sctx[r + 8 * half][h * HD_ + dt * 16 + cn] = (__bf16)acc[r];
    }
    __syncthreads();
  }

  // mean_attn out
  {
    float* map = out_ma + ((size_t)b * S_ + s0 + row16) * MEM_;
#pragma unroll
    for (int j = 0; j < 16; ++j) map[ln16 + 16 * j] = macc[j];
  }

  // stage 3: readout = concat(x, ctx) @ Wo + bo   (K = 512, pipelined)
  for (int ct = wave; ct < 16; ct += 8) {
    int colg = ct * 16 + cn;
    float bias = bo[colg];
    v8f acc;
#pragma unroll
    for (int r = 0; r < 8; ++r) acc[r] = bias;
    v16bf bm = load_b_packed(Wo_p + (size_t)ct * TILE_E);
#pragma unroll
    for (int kk = 0; kk < 16; ++kk) {
      v16bf bnext = bm;
      if (kk < 15)
        bnext = load_b_packed(Wo_p + (size_t)((kk + 1) * 16 + ct) * TILE_E);
      v16bf a = (kk < 8) ? lds_load_a(&sx[0][0], LDP, kk * 32)
                         : lds_load_a(&sctx[0][0], LDP, (kk - 8) * 32);
      acc = wmma_bf16(a, bm, acc);
      bm = bnext;
    }
#pragma unroll
    for (int r = 0; r < 8; ++r) sf[r + 8 * half][colg] = acc[r];
  }
  __syncthreads();

  // stage 4: LayerNorm + gate
  {
    float vloc[16];
    float s1 = 0.f;
#pragma unroll
    for (int j = 0; j < 16; ++j) {
      vloc[j] = sf[row16][ln16 + 16 * j];
      s1 += vloc[j];
    }
    for (int m = 1; m < 16; m <<= 1) s1 += __shfl_xor(s1, m, 32);
    float mu = s1 * (1.f / 256.f);
    float s2 = 0.f;
#pragma unroll
    for (int j = 0; j < 16; ++j) {
      float d = vloc[j] - mu;
      s2 += d * d;
    }
    for (int m = 1; m < 16; m <<= 1) s2 += __shfl_xor(s2, m, 32);
    float rstd = rsqrtf(s2 * (1.f / 256.f) + 1e-5f);
    float gdot = 0.f;
    float* rop = out_ro + ((size_t)b * S_ + s0 + row16) * H_;
#pragma unroll
    for (int j = 0; j < 16; ++j) {
      int c = ln16 + 16 * j;
      float y = (vloc[j] - mu) * rstd * ln_g[c] + ln_b[c];
      rop[c] = y;
      gdot += (float)sx[row16][c] * Wg[c] + y * Wg[H_ + c];
    }
    for (int m = 1; m < 16; m <<= 1) gdot += __shfl_xor(gdot, m, 32);
    if (ln16 == 0) srow[row16] = 1.f / (1.f + __expf(-(gdot + bg[0])));
  }
  __syncthreads();

  // stage 5: gsum[b,c] += sum_s gate[s]*x[s,c]
  {
    int c = tid;
    float acc = 0.f;
#pragma unroll
    for (int r = 0; r < 16; ++r) acc += srow[r] * (float)sx[r][c];
    atomicAdd(&gsum[b * H_ + c], acc);
  }
}

__global__ __launch_bounds__(256) void ww_kernel(const float* __restrict__ avg,
                                                 const float* __restrict__ We,
                                                 const float* __restrict__ be,
                                                 float* __restrict__ ww_out,
                                                 float* __restrict__ ww_ws) {
  __shared__ float red[256];
  __shared__ float ain[256];
  int b = blockIdx.x, m = threadIdx.x;
  ain[m] = avg[b * H_ + m];
  __syncthreads();
  float acc = be[m];
  for (int k = 0; k < H_; ++k) acc += ain[k] * We[k * MEM_ + m];
  red[m] = acc;
  __syncthreads();
  for (int s = 128; s > 0; s >>= 1) {
    if (m < s) red[m] = fmaxf(red[m], red[m + s]);
    __syncthreads();
  }
  float mx = red[0];
  __syncthreads();
  float e = __expf(acc - mx);
  red[m] = e;
  __syncthreads();
  for (int s = 128; s > 0; s >>= 1) {
    if (m < s) red[m] += red[m + s];
    __syncthreads();
  }
  float w = e / red[0];
  ww_out[b * MEM_ + m] = w;
  ww_ws[b * MEM_ + m] = w;
}

__global__ __launch_bounds__(256) void newmem_kernel(
    const float* __restrict__ memory, const float* __restrict__ decay,
    const float* __restrict__ ww, const float* __restrict__ gsum,
    float* __restrict__ out_nm) {
  __shared__ float red[256];
  int b = blockIdx.y, m = blockIdx.x, h = threadIdx.x;
  size_t base = ((size_t)b * MEM_ + m) * H_;
  // decay (1,1,MEM) broadcasts over LAST axis of memory [B,MEM,H] -> indexes h
  float val = memory[base + h] * decay[h] + ww[b * MEM_ + m] * gsum[b * H_ + h];
  red[h] = val * val;
  __syncthreads();
  for (int s = 128; s > 0; s >>= 1) {
    if (h < s) red[h] += red[h + s];
    __syncthreads();
  }
  float nrm = sqrtf(red[0]);
  float sc = 16.f / fmaxf(nrm, 1e-12f);  // sqrt(H) = 16
  out_nm[base + h] = val * sc;
}

extern "C" void kernel_launch(void* const* d_in, const int* in_sizes, int n_in,
                              void* d_out, int out_size, void* d_ws, size_t ws_size,
                              hipStream_t stream) {
  const float* x = (const float*)d_in[0];
  const float* memory = (const float*)d_in[1];
  const float* Wq = (const float*)d_in[2];
  const float* bq = (const float*)d_in[3];
  const float* Wk = (const float*)d_in[4];
  const float* bk = (const float*)d_in[5];
  const float* Wv = (const float*)d_in[6];
  const float* bv = (const float*)d_in[7];
  const float* Wg = (const float*)d_in[8];
  const float* bg = (const float*)d_in[9];
  const float* We = (const float*)d_in[10];
  const float* be = (const float*)d_in[11];
  const float* Wo = (const float*)d_in[12];
  const float* bo = (const float*)d_in[13];
  const float* decay = (const float*)d_in[14];
  const float* ln_g = (const float*)d_in[15];
  const float* ln_b = (const float*)d_in[16];

  float* out = (float*)d_out;
  float* out_ro = out;
  float* out_ma = out + (size_t)B_ * S_ * H_;
  float* out_nm = out_ma + (size_t)B_ * S_ * MEM_;
  float* out_ww = out_nm + (size_t)B_ * MEM_ * H_;

  char* ws = (char*)d_ws;
  float* gsum = (float*)(ws);                      // 32 KB
  float* avg = (float*)(ws + 32768);               // 32 KB
  float* ww_ws = (float*)(ws + 65536);             // 32 KB
  __bf16* Wq_p = (__bf16*)(ws + 98304);            // 128 KB
  __bf16* Wk_p = (__bf16*)(ws + 98304 + 131072);   // 128 KB
  __bf16* Wv_p = (__bf16*)(ws + 98304 + 262144);   // 128 KB
  __bf16* Wo_p = (__bf16*)(ws + 98304 + 393216);   // 256 KB
  __bf16* Ktp = (__bf16*)(ws + 98304 + 655360);    // 4 MB packed
  __bf16* Vbp = (__bf16*)(ws + 98304 + 655360 + 4194304);  // 4 MB packed

  zero_kernel<<<(B_ * H_ + 255) / 256, 256, 0, stream>>>(gsum, B_ * H_);
  pack_weights_kernel<<<(20480 + 255) / 256, 256, 0, stream>>>(
      Wq, Wk, Wv, Wo, Wq_p, Wk_p, Wv_p, Wo_p);
  kv_kernel<<<B_ * (MEM_ / 16), 256, 0, stream>>>(memory, bk, bv, Wk_p, Wv_p, Ktp,
                                                  Vbp);
  xmean_kernel<<<B_, 256, 0, stream>>>(x, avg);
  attn_kernel<<<dim3(S_ / 16, B_), 256, 0, stream>>>(x, bq, bo, Wg, bg, ln_g, ln_b,
                                                     Wq_p, Wo_p, Ktp, Vbp, out_ro,
                                                     out_ma, gsum);
  ww_kernel<<<B_, 256, 0, stream>>>(avg, We, be, out_ww, ww_ws);
  newmem_kernel<<<dim3(MEM_, B_), 256, 0, stream>>>(memory, decay, ww_ws, gsum,
                                                    out_nm);
}
